// MultiHeadedAttention_31095563223736
// MI455X (gfx1250) — matrix operand order repaired and checked by reading, not compiled, not run on hardware
//
#include <hip/hip_runtime.h>

typedef __bf16 bf16;
typedef __attribute__((ext_vector_type(16))) __bf16 v16bf;
typedef __attribute__((ext_vector_type(8)))  __bf16 v8bf;
typedef __attribute__((ext_vector_type(8)))  float  v8f;
typedef unsigned short u16;
typedef unsigned int   u32;

// ---------- helpers ----------

__device__ __forceinline__ u16 f2bf(float f) {
  u32 u = __float_as_uint(f);
  u32 r = (u + 0x7FFFu + ((u >> 16) & 1u)) >> 16;   // round-to-nearest-even
  return (u16)r;
}

// Build a 16-element bf16 fragment from two contiguous 8-element (16B) runs.
__device__ __forceinline__ v16bf make_frag(const u16* lo, const u16* hi) {
  union { v16bf v; v8bf h[2]; } u;
  u.h[0] = *reinterpret_cast<const v8bf*>(lo);
  u.h[1] = *reinterpret_cast<const v8bf*>(hi);
  return u.v;
}

__device__ __forceinline__ v8f wmma_bf16(v16bf a, v16bf b, v8f c) {
  return __builtin_amdgcn_wmma_f32_16x16x32_bf16(
      /*neg_a=*/false, a, /*neg_b=*/false, b,
      /*c_mod=*/(short)0, c, /*reuse_a=*/false, /*reuse_b=*/false);
}

// ---------- elementwise fp32 -> bf16 ----------

__global__ void __launch_bounds__(256)
cvt_f32_bf16(const float* __restrict__ in, u16* __restrict__ out, int n) {
  int i = blockIdx.x * 256 + threadIdx.x;
  if (i < n) out[i] = f2bf(in[i]);
}

// ---------- W [K,N] fp32 -> WT [N,K] bf16 (tiled transpose) ----------

__global__ void __launch_bounds__(256)
transpose_to_bf16(const float* __restrict__ W, u16* __restrict__ WT, int Dd) {
  __shared__ float tile[32][33];
  int tx = threadIdx.x, ty = threadIdx.y;       // block (32, 8)
  int bx = blockIdx.x << 5, by = blockIdx.y << 5;
#pragma unroll
  for (int i = 0; i < 4; ++i)
    tile[ty + i * 8][tx] = W[(size_t)(by + ty + i * 8) * Dd + bx + tx];
  __syncthreads();
#pragma unroll
  for (int i = 0; i < 4; ++i)
    WT[(size_t)(bx + ty + i * 8) * Dd + by + tx] = f2bf(tile[tx][ty + i * 8]);
}

// ---------- WMMA GEMM: Y[M,N] = A[M,K](bf16) * BT[N,K]^T (bf16) + bias ----------
// 32x64 tile per wave. K loop manually unrolled 2x with alternating register
// buffers: no rotation copies, loads for one half-step overlap the other
// half-step's 8 WMMAs. Requires K % 64 == 0.
// mode 0: bf16 out, row-major [M,N]
// mode 1: bf16 out, transposed per-head: VT[((b*H+h)*64+dk)*S + s]
// mode 2: f32 out, row-major [M,N]
__global__ void __launch_bounds__(32)
gemm_bf16_wmma(const u16* __restrict__ A, const u16* __restrict__ BT,
               const float* __restrict__ bias,
               u16* __restrict__ outB, float* __restrict__ outF,
               int M, int N, int K, int mode, int Sdim, int Hheads) {
  int lane  = threadIdx.x & 31;
  int half  = lane >> 4;
  int mrow  = lane & 15;
  int nlane = lane & 15;
  int koffA = half * 16;
  int koffB = half * 16;
  int mbase = blockIdx.y << 5;   // 32 rows per wave
  int nbase = blockIdx.x << 6;   // 64 cols per wave

  const u16* arow[2];
  arow[0] = A + (size_t)(mbase + mrow) * K;
  arow[1] = A + (size_t)(mbase + 16 + mrow) * K;
  const u16* brow[4];
#pragma unroll
  for (int j = 0; j < 4; ++j)
    brow[j] = BT + (size_t)(nbase + j * 16 + nlane) * K;

  v8f acc[2][4] = {};
  v16bf a0[2], b0[4], a1[2], b1[4];

  auto loadA = [&](int kb, v16bf a[2]) {
#pragma unroll
    for (int mi = 0; mi < 2; ++mi)
      a[mi] = make_frag(arow[mi] + kb + koffA, arow[mi] + kb + koffA + 8);
  };
  auto loadB = [&](int kb, v16bf bf[4]) {
#pragma unroll
    for (int j = 0; j < 4; ++j)
      bf[j] = make_frag(brow[j] + kb + koffB, brow[j] + kb + koffB + 8);
  };
  auto mul8 = [&](const v16bf a[2], const v16bf bf[4]) {
#pragma unroll
    for (int mi = 0; mi < 2; ++mi)
#pragma unroll
      for (int j = 0; j < 4; ++j)
        acc[mi][j] = wmma_bf16(a[mi], bf[j], acc[mi][j]);
  };

  loadA(0, a0);
  loadB(0, b0);
  int kb = 0;
  for (; kb + 64 < K; kb += 64) {
    loadA(kb + 32, a1);
    loadB(kb + 32, b1);
    __builtin_prefetch(arow[0] + kb + 128, 0, 1);
    __builtin_prefetch(arow[1] + kb + 128, 0, 1);
    mul8(a0, b0);                 // consumes kb, overlaps kb+32 loads
    loadA(kb + 64, a0);
    loadB(kb + 64, b0);
    mul8(a1, b1);                 // consumes kb+32, overlaps kb+64 loads
  }
  loadA(kb + 32, a1);
  loadB(kb + 32, b1);
  mul8(a0, b0);
  mul8(a1, b1);

  int mtop = half * 8;
  float bv[4];
#pragma unroll
  for (int j = 0; j < 4; ++j)
    bv[j] = bias ? bias[nbase + j * 16 + nlane] : 0.0f;

#pragma unroll
  for (int mi = 0; mi < 2; ++mi)
#pragma unroll
    for (int j = 0; j < 4; ++j) {
      int n = nbase + j * 16 + nlane;
#pragma unroll
      for (int r = 0; r < 8; ++r) {
        float v = acc[mi][j][r] + bv[j];
        int m = mbase + mi * 16 + mtop + r;
        if (mode == 0) {
          outB[(size_t)m * N + n] = f2bf(v);
        } else if (mode == 1) {
          int b  = m / Sdim;
          int s  = m - b * Sdim;
          int h  = n >> 6;
          int dk = n & 63;
          outB[(((size_t)(b * Hheads + h) << 6) + dk) * Sdim + s] = f2bf(v);
        } else {
          outF[(size_t)m * N + n] = v;
        }
      }
    }
}

// ---------- flash attention per (b, h, 16-row block), single wave ----------

__global__ void __launch_bounds__(32)
attn_wmma(const u16* __restrict__ Qp, const u16* __restrict__ Kp,
          const u16* __restrict__ VTp, const int* __restrict__ mask,
          u16* __restrict__ ctx) {
  const int S = 2048, D = 1024, H = 16;
  int lane  = threadIdx.x & 31;
  int half  = lane >> 4;
  int mrow  = lane & 15;
  int nlane = lane & 15;
  int mtop  = half * 8;
  int ub    = blockIdx.x << 4;
  int h     = blockIdx.y;
  int b     = blockIdx.z;

  __shared__ __align__(16) u16 pls[16 * 32];   // P tile, row-major 16x32

  const u16* Qh   = Qp  + (size_t)b * S * D + h * 64;
  const u16* Kh   = Kp  + (size_t)b * S * D + h * 64;
  const u16* VTh  = VTp + ((size_t)(b * H + h) * 64) * S;
  const int* maskB = mask + (size_t)b * S * S;

  // Q fragments for this 16-row block (k = 0..31 and 32..63)
  const u16* qrow = Qh + (size_t)(ub + mrow) * D;
  v16bf aQ0 = make_frag(qrow + half * 16,      qrow + half * 16 + 8);
  v16bf aQ1 = make_frag(qrow + 32 + half * 16, qrow + 40 + half * 16);

  auto loadK = [&](int t, v16bf bK[2][2]) {
#pragma unroll
    for (int j = 0; j < 2; ++j) {
      const u16* krow = Kh + (size_t)(t + j * 16 + nlane) * D + half * 16;
      bK[j][0] = make_frag(krow,      krow + 8);
      bK[j][1] = make_frag(krow + 32, krow + 40);
    }
  };

  float Mrow[8], Lrow[8];
  v8f o[4] = {};
#pragma unroll
  for (int r = 0; r < 8; ++r) { Mrow[r] = -3.0e38f; Lrow[r] = 0.0f; }

  v16bf bK[2][2], bKn[2][2];
  loadK(0, bK);                       // prologue

  for (int t = 0; t < S; t += 32) {
    // ---- V fragments + mask for this block (consumed late -> overlap)
    v16bf bV[4];
    int   mk[2][8];
#pragma unroll
    for (int dkt = 0; dkt < 4; ++dkt) {
      const u16* vrow = VTh + (size_t)(dkt * 16 + nlane) * S + t + half * 16;
      bV[dkt] = make_frag(vrow, vrow + 8);
    }
#pragma unroll
    for (int j = 0; j < 2; ++j) {
      const int* mp = maskB + (size_t)(ub + mtop) * S + t + j * 16 + nlane;
#pragma unroll
      for (int r = 0; r < 8; ++r)
        mk[j][r] = mp[(size_t)r * S];
    }

    // ---- scores (consume bK)
    v8f sc[2];
#pragma unroll
    for (int j = 0; j < 2; ++j) {
      v8f s = {};
      s = wmma_bf16(aQ0, bK[j][0], s);
      s = wmma_bf16(aQ1, bK[j][1], s);
#pragma unroll
      for (int r = 0; r < 8; ++r) {
        float v = s[r] * 0.125f;                 // 1/sqrt(64)
        if (mk[j][r] == 0) v = -3.0e38f;
        s[r] = v;
      }
      sc[j] = s;
    }

    // ---- issue next block's K-fragment loads; softmax below hides them
    int tn = (t + 32 < S) ? (t + 32) : t;
    loadK(tn, bKn);

    // ---- online softmax (rows live one-per-lane across a 16-lane half)
    float nm[8], alpha[8];
#pragma unroll
    for (int r = 0; r < 8; ++r) {
      float vm = fmaxf(sc[0][r], sc[1][r]);
#pragma unroll
      for (int d = 1; d < 16; d <<= 1)
        vm = fmaxf(vm, __shfl_xor(vm, d, 32));
      nm[r]    = fmaxf(Mrow[r], vm);
      alpha[r] = __expf(Mrow[r] - nm[r]);
      Mrow[r]  = nm[r];
    }
#pragma unroll
    for (int j = 0; j < 2; ++j)
#pragma unroll
      for (int r = 0; r < 8; ++r)
        sc[j][r] = __expf(sc[j][r] - nm[r]);
#pragma unroll
    for (int r = 0; r < 8; ++r) {
      float rs = sc[0][r] + sc[1][r];
#pragma unroll
      for (int d = 1; d < 16; d <<= 1)
        rs += __shfl_xor(rs, d, 32);
      Lrow[r] = Lrow[r] * alpha[r] + rs;
    }
#pragma unroll
    for (int dkt = 0; dkt < 4; ++dkt)
#pragma unroll
      for (int r = 0; r < 8; ++r)
        o[dkt][r] *= alpha[r];

    // ---- C-layout -> A-layout for P via LDS
    __syncthreads();
#pragma unroll
    for (int j = 0; j < 2; ++j)
#pragma unroll
      for (int r = 0; r < 8; ++r)
        pls[(mtop + r) * 32 + j * 16 + nlane] = f2bf(sc[j][r]);
    __syncthreads();

    const u16* prow = pls + mrow * 32 + half * 16;
    v16bf aP = make_frag(prow, prow + 8);
#pragma unroll
    for (int dkt = 0; dkt < 4; ++dkt)
      o[dkt] = wmma_bf16(aP, bV[dkt], o[dkt]);

    // ---- rotate K double-buffer
#pragma unroll
    for (int j = 0; j < 2; ++j) {
      bK[j][0] = bKn[j][0];
      bK[j][1] = bKn[j][1];
    }
  }

  // normalize and emit context in [B,S,H*DK] layout (= transpose+reshape)
#pragma unroll
  for (int dkt = 0; dkt < 4; ++dkt)
#pragma unroll
    for (int r = 0; r < 8; ++r) {
      float v = o[dkt][r] / Lrow[r];
      ctx[((size_t)b * S + ub + mtop + r) * D + h * 64 + dkt * 16 + nlane] = f2bf(v);
    }
}

// ---------- host ----------

extern "C" void kernel_launch(void* const* d_in, const int* in_sizes, int n_in,
                              void* d_out, int out_size, void* d_ws, size_t ws_size,
                              hipStream_t stream) {
  (void)in_sizes; (void)n_in; (void)out_size; (void)ws_size;
  const int Bb = 4, Ss = 2048, Dd = 1024, Hh = 16;
  const int Mm = Bb * Ss;                 // 8192
  const size_t MS = (size_t)Mm * Dd;      // 8,388,608
  const size_t WS = (size_t)Dd * Dd;      // 1,048,576

  const float* q    = (const float*)d_in[0];
  const float* k    = (const float*)d_in[1];
  const float* v    = (const float*)d_in[2];
  const int*   mask = (const int*)  d_in[3];
  const float* Wq   = (const float*)d_in[4];
  const float* bq   = (const float*)d_in[5];
  const float* Wk   = (const float*)d_in[6];
  const float* bk   = (const float*)d_in[7];
  const float* Wv   = (const float*)d_in[8];
  const float* bv   = (const float*)d_in[9];
  const float* Wo   = (const float*)d_in[10];
  const float* bo   = (const float*)d_in[11];

  char* ws = (char*)d_ws;
  u16* qbf = (u16*)ws; ws += MS * 2;
  u16* kbf = (u16*)ws; ws += MS * 2;
  u16* vbf = (u16*)ws; ws += MS * 2;
  u16* WqT = (u16*)ws; ws += WS * 2;
  u16* WkT = (u16*)ws; ws += WS * 2;
  u16* WvT = (u16*)ws; ws += WS * 2;
  u16* WoT = (u16*)ws; ws += WS * 2;
  u16* Qp  = (u16*)ws; ws += MS * 2;
  u16* Kp  = (u16*)ws; ws += MS * 2;
  u16* VTp = (u16*)ws; ws += MS * 2;
  u16* ctx = (u16*)ws; ws += MS * 2;

  int cvtBlocks = (int)((MS + 255) / 256);
  cvt_f32_bf16<<<cvtBlocks, 256, 0, stream>>>(q, qbf, (int)MS);
  cvt_f32_bf16<<<cvtBlocks, 256, 0, stream>>>(k, kbf, (int)MS);
  cvt_f32_bf16<<<cvtBlocks, 256, 0, stream>>>(v, vbf, (int)MS);

  dim3 tb(32, 8);
  dim3 tg(Dd / 32, Dd / 32);
  transpose_to_bf16<<<tg, tb, 0, stream>>>(Wq, WqT, Dd);
  transpose_to_bf16<<<tg, tb, 0, stream>>>(Wk, WkT, Dd);
  transpose_to_bf16<<<tg, tb, 0, stream>>>(Wv, WvT, Dd);
  transpose_to_bf16<<<tg, tb, 0, stream>>>(Wo, WoT, Dd);

  dim3 gg(Dd / 64, Mm / 32);   // (16, 256) waves, 32x64 tile each
  gemm_bf16_wmma<<<gg, 32, 0, stream>>>(qbf, WqT, bq, Qp,  nullptr, Mm, Dd, Dd, 0, Ss, Hh);
  gemm_bf16_wmma<<<gg, 32, 0, stream>>>(kbf, WkT, bk, Kp,  nullptr, Mm, Dd, Dd, 0, Ss, Hh);
  gemm_bf16_wmma<<<gg, 32, 0, stream>>>(vbf, WvT, bv, VTp, nullptr, Mm, Dd, Dd, 1, Ss, Hh);

  dim3 ag(Ss / 16, Hh, Bb);
  attn_wmma<<<ag, 32, 0, stream>>>(Qp, Kp, VTp, mask, ctx);

  gemm_bf16_wmma<<<gg, 32, 0, stream>>>(ctx, WoT, bo, nullptr, (float*)d_out, Mm, Dd, Dd, 2, Ss, Hh);
}
